// LinkPredictionGNN_18648747999233
// MI455X (gfx1250) — matrix operand order repaired and checked by reading, not compiled, not run on hardware
//
#include <hip/hip_runtime.h>
#include <stdint.h>

typedef __attribute__((ext_vector_type(16))) __bf16 v16bf;
typedef __attribute__((ext_vector_type(8)))  float  v8f;

// Hardware FP32 global atomic add (no-return form -> STOREcnt, no CAS loop).
__device__ __forceinline__ void gatomic_add_f32(float* p, float v) {
  asm volatile("global_atomic_add_f32 %0, %1, off" :: "v"(p), "v"(v) : "memory");
}

__device__ __forceinline__ uint16_t f32_to_bf16_rne(float f) {
  uint32_t b = __float_as_uint(f);
  return (uint16_t)((b + 0x7FFFu + ((b >> 16) & 1u)) >> 16);
}

__global__ void k_f32_to_bf16(const float* __restrict__ in, uint16_t* __restrict__ out,
                              long long n) {
  long long i = (long long)blockIdx.x * blockDim.x + threadIdx.x;
  if (i >= n) return;
  out[i] = f32_to_bf16_rne(in[i]);
}

// Pack weight W[K x N] (f32, row-major) into WMMA-B fragment layout, bf16:
// Bp element j: i = j&15, lane = (j>>4)&31, block = j>>9; block = tn*KB + kkIdx.
// Fragment element i of lane = B[kkIdx*32 + (lane>>4)*16 + i][tn*16 + (lane&15)].
__global__ void k_pack_b(const float* __restrict__ W, uint16_t* __restrict__ Bp,
                         int N, int KB, long long total) {
  long long j = (long long)blockIdx.x * blockDim.x + threadIdx.x;
  if (j >= total) return;
  int i     = (int)(j & 15);
  int lane  = (int)((j >> 4) & 31);
  int block = (int)(j >> 9);
  int kkIdx = block % KB;
  int tn    = block / KB;
  int k = kkIdx * 32 + (lane >> 4) * 16 + i;
  int n = tn * 16 + (lane & 15);
  Bp[j] = f32_to_bf16_rne(W[(long long)k * N + n]);
}

__global__ void k_fill_f32(float* __restrict__ p, float v, int n) {
  int i = blockIdx.x * blockDim.x + threadIdx.x;
  if (i < n) p[i] = v;
}

__global__ void k_deg_scatter(const int* __restrict__ dst, float* __restrict__ deg, int E) {
  int e = blockIdx.x * blockDim.x + threadIdx.x;
  if (e < E) gatomic_add_f32(&deg[dst[e]], 1.0f);
}

__global__ void k_rsqrt_inplace(float* __restrict__ p, int n) {
  int i = blockIdx.x * blockDim.x + threadIdx.x;
  if (i < n) p[i] = rsqrtf(p[i]);  // deg >= 1 always (self-loop)
}

// H[MxN] = A[MxK](bf16,row-major) * B(packed fragments), f32 accumulate.
// KB = K/32 (K=128 -> 4). Each wave owns one column tile tn, loads all KB B
// fragments ONCE into registers, then streams MT m-tiles of A (b128 loads).
// WITH_AGG: also write AGG = dinv[row]^2 * h (+ bias[col] if WITH_BIAS) —
// compile-time so the epilogue has no per-element branches.
// M%16==0 (50000 ok); EXEC all-1s (guards are wave-uniform).
template <int KB, bool WITH_AGG, bool WITH_BIAS>
__global__ void k_gemm_bf16_wmma(const uint16_t* __restrict__ A, const uint16_t* __restrict__ Bp,
                                 float* __restrict__ H, float* __restrict__ AGG,
                                 const float* __restrict__ dinv, const float* __restrict__ bias,
                                 int M, int N, int mt_per_wave) {
  const int K = KB * 32;
  int wave   = blockIdx.x * (blockDim.x >> 5) + (threadIdx.x >> 5);
  int ntn    = N >> 4;
  int mtiles = M >> 4;
  int tn  = wave % ntn;
  int mt0 = (wave / ntn) * mt_per_wave;
  if (mt0 >= mtiles) return;            // wave-uniform
  int lane = threadIdx.x & 31;
  int l15  = lane & 15;
  int half = lane >> 4;

  // Load all B fragments once: 2 x b128 per fragment, coalesced (32B/lane).
  union bfrag { v16bf v; uint4 q[2]; };
  bfrag b[KB];
  const uint16_t* bp = Bp + ((size_t)tn * KB * 32 + lane) * 16;
#pragma unroll
  for (int kb = 0; kb < KB; ++kb) {
    const uint16_t* p = bp + (size_t)kb * 32 * 16;
    b[kb].q[0] = *(const uint4*)(p);
    b[kb].q[1] = *(const uint4*)(p + 8);
  }
  float bias_v = WITH_BIAS ? bias[(tn << 4) + l15] : 0.0f;

  for (int t = 0; t < mt_per_wave; ++t) {
    int tm = mt0 + t;
    if (tm >= mtiles) break;            // wave-uniform
    int row = (tm << 4) + l15;
    const uint16_t* arow = A + (size_t)row * K;
    __builtin_prefetch(arow + (size_t)16 * K, 0, 1);  // next m-tile stream
    v8f acc = {};
#pragma unroll
    for (int kb = 0; kb < KB; ++kb) {
      // A fragment (16x32): half 0 -> K {kb*32+0..15}; half 1 -> {+16..31}
      union { v16bf v; uint4 q[2]; } a;
      const uint16_t* ap = arow + kb * 32 + half * 16;
      a.q[0] = *(const uint4*)(ap);
      a.q[1] = *(const uint4*)(ap + 8);
      acc = __builtin_amdgcn_wmma_f32_16x16x32_bf16(
          false, a.v, false, b[kb].v, (short)0, acc, false, false);
    }
    // C/D layout: VGPR r -> row (tm*16 + r + 8*half), col = tn*16 + l15
    int rowBase = (tm << 4) + half * 8;
    size_t cOff = (size_t)rowBase * N + (tn << 4) + l15;
    float* hp = H + cOff;
#pragma unroll
    for (int r = 0; r < 8; ++r) hp[(size_t)r * N] = acc[r];
    if (WITH_AGG) {
      float d2[8];
#pragma unroll
      for (int r = 0; r < 8; ++r) { float di = dinv[rowBase + r]; d2[r] = di * di; }
      float* gp = AGG + cOff;
#pragma unroll
      for (int r = 0; r < 8; ++r) gp[(size_t)r * N] = acc[r] * d2[r] + bias_v;
    }
  }
}

// agg[dst,c] += dinv[src]*dinv[dst] * h[src,c] via hw fp32 atomics (L2-resident).
template <int C>
__global__ void k_edge_scatter(const float* __restrict__ h, const int* __restrict__ src,
                               const int* __restrict__ dst, const float* __restrict__ dinv,
                               float* __restrict__ agg, long long total) {
  long long i = (long long)blockIdx.x * blockDim.x + threadIdx.x;
  if (i >= total) return;
  long long e = i / C;
  int c = (int)(i & (C - 1));
  int s = src[e], d = dst[e];
  float nrm = dinv[s] * dinv[d];
  gatomic_add_f32(agg + (long long)d * C + c, h[(long long)s * C + c] * nrm);
}

// xb = bf16(relu(a + bias)) : fused layer-1 activation + cast for layer-2 GEMM input.
__global__ void k_bias_relu_bf16(const float* __restrict__ a, const float* __restrict__ b,
                                 uint16_t* __restrict__ xb, long long total, int cmask) {
  long long i = (long long)blockIdx.x * blockDim.x + threadIdx.x;
  if (i >= total) return;
  xb[i] = f32_to_bf16_rne(fmaxf(a[i] + b[i & cmask], 0.0f));
}

extern "C" void kernel_launch(void* const* d_in, const int* in_sizes, int n_in,
                              void* d_out, int out_size, void* d_ws, size_t ws_size,
                              hipStream_t stream) {
  (void)n_in; (void)out_size; (void)ws_size;
  const float* x  = (const float*)d_in[0];
  const int*   ei = (const int*)d_in[1];
  const float* W1 = (const float*)d_in[2];
  const float* b1 = (const float*)d_in[3];
  const float* W2 = (const float*)d_in[4];
  const float* b2 = (const float*)d_in[5];

  const int C1 = in_sizes[3];        // 128 (hidden)
  const int C2 = in_sizes[5];        // 64  (out)
  const int IN = in_sizes[2] / C1;   // 128 (in)
  const int N  = in_sizes[0] / IN;   // 50000
  const int E  = in_sizes[1] / 2;    // 800000
  const int* src = ei;
  const int* dst = ei + E;

  // Workspace carve-up (256B aligned). xb reused for layer-2 input; h1 reused as h2.
  char* w = (char*)d_ws;
  auto al = [](size_t o) { return (o + 255) & ~(size_t)255; };
  size_t o = 0;
  float*    dinv = (float*)(w + o);    o = al(o + (size_t)N * 4);
  uint16_t* xb   = (uint16_t*)(w + o); o = al(o + (size_t)N * IN * 2);
  uint16_t* B1p  = (uint16_t*)(w + o); o = al(o + (size_t)IN * C1 * 2);
  uint16_t* B2p  = (uint16_t*)(w + o); o = al(o + (size_t)C1 * C2 * 2);
  float*    h1   = (float*)(w + o);    o = al(o + (size_t)N * C1 * 4);
  float*    agg1 = (float*)(w + o);    o = al(o + (size_t)N * C1 * 4);
  float*    h2   = h1;
  float*    out  = (float*)d_out;

  const int TB = 256;                       // 8 waves/block
  auto nb = [&](long long tot) { return (unsigned)((tot + TB - 1) / TB); };

  // --- precompute: bf16 cast of x, fragment-packed weights, symmetric-norm dinv ---
  long long nx = (long long)N * IN;
  hipLaunchKernelGGL(k_f32_to_bf16, dim3(nb(nx)), dim3(TB), 0, stream, x, xb, nx);
  hipLaunchKernelGGL(k_pack_b, dim3(nb((long long)IN * C1)), dim3(TB), 0, stream,
                     W1, B1p, C1, IN / 32, (long long)IN * C1);
  hipLaunchKernelGGL(k_pack_b, dim3(nb((long long)C1 * C2)), dim3(TB), 0, stream,
                     W2, B2p, C2, C1 / 32, (long long)C1 * C2);
  hipLaunchKernelGGL(k_fill_f32, dim3(nb(N)), dim3(TB), 0, stream, dinv, 1.0f, N);
  hipLaunchKernelGGL(k_deg_scatter, dim3(nb(E)), dim3(TB), 0, stream, dst, dinv, E);
  hipLaunchKernelGGL(k_rsqrt_inplace, dim3(nb(N)), dim3(TB), 0, stream, dinv, N);

  const int MT_PER_WAVE = 8;
  const int WPB = TB >> 5;

  // --- layer 1: h1 = x@W1 (WMMA, epilogue agg1 = dinv^2*h1) ; edge scatter ; relu+cast ---
  {
    int mchunks = (N / 16 + MT_PER_WAVE - 1) / MT_PER_WAVE;
    int waves = mchunks * (C1 >> 4);
    hipLaunchKernelGGL((k_gemm_bf16_wmma<4, true, false>),
                       dim3((waves + WPB - 1) / WPB), dim3(TB), 0, stream,
                       xb, B1p, h1, agg1, dinv, (const float*)nullptr, N, C1, MT_PER_WAVE);
  }
  long long e1 = (long long)E * C1;
  hipLaunchKernelGGL(k_edge_scatter<128>, dim3(nb(e1)), dim3(TB), 0, stream,
                     h1, src, dst, dinv, agg1, e1);
  long long t1 = (long long)N * C1;
  hipLaunchKernelGGL(k_bias_relu_bf16, dim3(nb(t1)), dim3(TB), 0, stream,
                     agg1, b1, xb, t1, C1 - 1);

  // --- layer 2: h2 = relu1@W2 (WMMA, epilogue out = dinv^2*h2 + b2) ; edge scatter ---
  {
    int mchunks = (N / 16 + MT_PER_WAVE - 1) / MT_PER_WAVE;
    int waves = mchunks * (C2 >> 4);
    hipLaunchKernelGGL((k_gemm_bf16_wmma<4, true, true>),
                       dim3((waves + WPB - 1) / WPB), dim3(TB), 0, stream,
                       xb, B2p, h2, out, dinv, b2, N, C2, MT_PER_WAVE);
  }
  long long e2 = (long long)E * C2;
  hipLaunchKernelGGL(k_edge_scatter<64>, dim3(nb(e2)), dim3(TB), 0, stream,
                     h2, src, dst, dinv, out, e2);
}
